// MultiHeadAttention_73967926772025
// MI455X (gfx1250) — matrix operand order repaired and checked by reading, not compiled, not run on hardware
//
#include <hip/hip_runtime.h>

// ---------------------------------------------------------------------------
// MHA forward for MI455X (gfx1250): f16 WMMA GEMMs with f32 accumulation,
// TDM (tensor_load_to_lds) double-buffered B-tile staging for weight GEMMs.
// Shapes: B=8, G=4, L=512, D_MODEL=D_K=D_V=512, H=8, d_head=64.
// Outputs: y [B,G,L,512] f32, then attn [H*B,G,L,L] f32 (concatenated).
// ---------------------------------------------------------------------------

typedef __attribute__((ext_vector_type(16))) _Float16 v16h;
typedef __attribute__((ext_vector_type(8)))  float    v8f;
typedef __attribute__((ext_vector_type(4)))  unsigned int u32x4;
typedef __attribute__((ext_vector_type(8)))  int      i32x8;
typedef __attribute__((ext_vector_type(4)))  int      i32x4;

#define DM    512
#define HEADS 8
#define DH    64
#define BB    8
#define GG    4
#define LL    512
#define ROWS  (BB * GG * LL)   // 16384

#if defined(__gfx1250__) && __has_builtin(__builtin_amdgcn_tensor_load_to_lds) \
    && __has_builtin(__builtin_amdgcn_s_wait_tensorcnt)
#define USE_TDM 1
#else
#define USE_TDM 0
#endif

// ---- WMMA fragment helpers -------------------------------------------------
// A (16x32 f16, MxK): lane l holds row M=l&15; halves 0..7 are K = kbase..+7,
// halves 8..15 are K = kbase+16..+23, kbase = (l>>4)*8.  => two 16B runs.
// B (32x16 f16, KxN) stored N-major [n][k]: lane l holds col N=l&15;
// halves 0..7 are K = kbase..+7, halves 8..15 are K = kbase+16..+23,
// kbase = (l>>4)*8 (same as A).                          => two 16B runs.
// C/D (16x16 f32): vgpr r, lane l -> M = r + (l>>4)*8, N = l&15.

__device__ __forceinline__ v16h load_a_f32(const float* p, int ld, int row0,
                                           int k0, int lane) {
  const float* src = p + (size_t)(row0 + (lane & 15)) * ld +
                     (k0 + ((lane >> 4) << 3));
  v16h a;
#pragma unroll
  for (int i = 0; i < 8; ++i) {
    a[i]     = (_Float16)src[i];
    a[i + 8] = (_Float16)src[i + 16];
  }
  return a;
}

__device__ __forceinline__ v16h load_a_f16(const _Float16* p, int ld, int row0,
                                           int k0, int lane) {
  const _Float16* src = p + (size_t)(row0 + (lane & 15)) * ld +
                        (k0 + ((lane >> 4) << 3));
  v16h a;
#pragma unroll
  for (int i = 0; i < 8; ++i) {
    a[i]     = src[i];
    a[i + 8] = src[i + 16];
  }
  return a;
}

__device__ __forceinline__ v16h load_b_f16(const _Float16* p, int ld, int n0,
                                           int k0, int lane) {
  const _Float16* src = p + (size_t)(n0 + (lane & 15)) * ld +
                        (k0 + ((lane >> 4) << 3));
  v16h b;
#pragma unroll
  for (int i = 0; i < 8; ++i) {
    b[i]     = src[i];
    b[i + 8] = src[i + 16];
  }
  return b;
}

__device__ __forceinline__ v8f wmma16(v16h a, v16h b, v8f c) {
  return __builtin_amdgcn_wmma_f32_16x16x32_f16(false, a, false, b, (short)0,
                                                c, false, false);
}

__device__ __forceinline__ float wave_sum(float x) {
#pragma unroll
  for (int off = 16; off >= 1; off >>= 1) x += __shfl_xor(x, off, 32);
  return x;
}

__device__ __forceinline__ float wave_max(float x) {
#pragma unroll
  for (int off = 16; off >= 1; off >>= 1) x = fmaxf(x, __shfl_xor(x, off, 32));
  return x;
}

#if USE_TDM
// ---- TDM: DMA a 64(n) x 32(k) f16 tile of an N-major [512][512] matrix
// into LDS at lds_byte_off. D# per CDNA5 ISA.
__device__ __forceinline__ void tdm_load_tile_64x32(const _Float16* gsrc,
                                                    unsigned lds_byte_off) {
  unsigned long long ga = (unsigned long long)(size_t)gsrc;
  u32x4 g0;
  g0[0] = 1u;                                   // count=1, user descriptor
  g0[1] = lds_byte_off;                         // lds_addr (bytes)
  g0[2] = (unsigned)(ga & 0xFFFFFFFFu);         // global_addr[31:0]
  g0[3] = (unsigned)((ga >> 32) & 0x01FFFFFFu)  // global_addr[56:32]
          | (2u << 30);                         // type = 2 ("image")
  i32x8 g1;
  g1[0] = 0x00010000;        // workgroup_mask=0, data_size=1 (2 bytes)
  g1[1] = (int)(512u << 16); // tensor_dim0 (k) = 512  (bits 79:48, lo16)
  g1[2] = (int)(512u << 16); // dim0 hi=0 | tensor_dim1 (n) = 512 lo16
  g1[3] = (int)(32u << 16);  // dim1 hi=0 | tile_dim0 (k) = 32
  g1[4] = 64;                // tile_dim1 (n rows) = 64, tile_dim2 = 0
  g1[5] = 512;               // tensor_dim0_stride = 512 elems (lo32)
  g1[6] = 0;                 // stride0 hi / stride1 lo (unused for 2D)
  g1[7] = 0;
  i32x4 z4 = {0, 0, 0, 0};
#if __has_include(<hip/amd_detail/amd_gfx1250_TDM.h>)
  i32x8 z8 = {0, 0, 0, 0, 0, 0, 0, 0};
  __builtin_amdgcn_tensor_load_to_lds(g0, g1, z4, z4, z8, 0);
#else
  __builtin_amdgcn_tensor_load_to_lds(g0, g1, z4, z4, 0);
#endif
}
#endif  // USE_TDM

// ---- Kernel 0: convert + transpose weights to f16 [n][k] -------------------
__global__ __launch_bounds__(256) void wconv_kernel(
    const float* __restrict__ Wq, const float* __restrict__ Wk,
    const float* __restrict__ Wv, const float* __restrict__ Wfc,
    _Float16* __restrict__ Wt) {
  int idx = blockIdx.x * 256 + threadIdx.x;     // 4 * 512 * 512 total
  int w = idx >> 18;
  int r = idx & 262143;
  int n = r >> 9;
  int k = r & 511;
  const float* W = (w == 0) ? Wq : (w == 1) ? Wk : (w == 2) ? Wv : Wfc;
  Wt[idx] = (_Float16)W[k * DM + n];            // Wt[w][n][k] = W[k][n]
}

// ---- Kernel 1: QKV projections ---------------------------------------------
// grid = (ROWS/128, DM/64, 3). Block 256 thr = 8 waves; wave w: rows w*16..+16,
// cols n0..n0+63 (4 tiles). B tile (64n x 32k) staged in LDS via TDM,
// double buffered. Writes head-split f16: Q2/K2 [bgh][l][d], Vt [bgh][d][l].
__global__ __launch_bounds__(256) void qkv_gemm_kernel(
    const float* __restrict__ q, const float* __restrict__ k,
    const float* __restrict__ v, const _Float16* __restrict__ Wt,
    const float* __restrict__ bq, const float* __restrict__ bk,
    const float* __restrict__ bv, _Float16* __restrict__ Q2,
    _Float16* __restrict__ K2, _Float16* __restrict__ Vt) {
  const int which = blockIdx.z;
  const float* X = (which == 0) ? q : (which == 1) ? k : v;
  const _Float16* Wm = Wt + (size_t)which * DM * DM;
  const float* bias = (which == 0) ? bq : (which == 1) ? bk : bv;

  const int wave = threadIdx.x >> 5, lane = threadIdx.x & 31;
  const int row0 = blockIdx.x * 128 + wave * 16;
  const int n0 = blockIdx.y * 64;

  v8f acc[4] = {v8f{}, v8f{}, v8f{}, v8f{}};

#if USE_TDM
  __shared__ _Float16 Bt[2][64 * 32];   // double-buffered B tile (8 KB)
  if (threadIdx.x < 32)
    tdm_load_tile_64x32(Wm + (size_t)n0 * DM, (unsigned)(size_t)&Bt[0][0]);
  for (int step = 0; step < 16; ++step) {
    const int kb = step * 32;
    __syncthreads();  // everyone done reading buffer (step+1)&1 from step-1
    if (threadIdx.x < 32 && step + 1 < 16)
      tdm_load_tile_64x32(Wm + (size_t)n0 * DM + (kb + 32),
                          (unsigned)(size_t)&Bt[(step + 1) & 1][0]);
    if (step + 1 < 16)
      __builtin_amdgcn_s_wait_tensorcnt(1);   // current buffer complete
    else
      __builtin_amdgcn_s_wait_tensorcnt(0);
    __syncthreads();
    const _Float16* Bcur = &Bt[step & 1][0];
    v16h a = load_a_f32(X, DM, row0, kb, lane);
#pragma unroll
    for (int nt = 0; nt < 4; ++nt) {
      v16h b = load_b_f16(Bcur, 32, nt * 16, 0, lane);
      acc[nt] = wmma16(a, b, acc[nt]);
    }
  }
#else
  for (int kb = 0; kb < DM; kb += 32) {
    v16h a = load_a_f32(X, DM, row0, kb, lane);
#pragma unroll
    for (int nt = 0; nt < 4; ++nt) {
      v16h b = load_b_f16(Wm, DM, n0 + nt * 16, kb, lane);
      acc[nt] = wmma16(a, b, acc[nt]);
    }
  }
#endif

#pragma unroll
  for (int nt = 0; nt < 4; ++nt) {
#pragma unroll
    for (int r = 0; r < 8; ++r) {
      int m = row0 + r + ((lane >> 4) << 3);
      int c = n0 + nt * 16 + (lane & 15);
      float val = acc[nt][r] + bias[c];
      int bg = m >> 9, l = m & 511;
      int h = c >> 6, d = c & 63;
      int bgh = bg * HEADS + h;
      _Float16 hv = (_Float16)val;
      if (which == 0)
        Q2[((size_t)bgh * LL + l) * DH + d] = hv;
      else if (which == 1)
        K2[((size_t)bgh * LL + l) * DH + d] = hv;
      else
        Vt[((size_t)bgh * DH + d) * LL + l] = hv;
    }
  }
}

// ---- Kernel 2: attention (scores -> softmax -> attn out -> P*V) ------------
// grid = (B*G*H, L/16). Block 256 thr = 8 waves; 16 query rows per block.
__global__ __launch_bounds__(256) void attn_kernel(
    const _Float16* __restrict__ Q2, const _Float16* __restrict__ K2,
    const _Float16* __restrict__ Vt, float* __restrict__ attn_out,
    _Float16* __restrict__ Oc) {
  __shared__ float S[16][516];        // raw scores (padded)
  __shared__ _Float16 P[16][520];     // softmax probs, f16 (padded)
  __shared__ float Op[2][16][64];     // partial outputs (K halves)

  const int bgh = blockIdx.x;
  const int qBase = blockIdx.y * 16;
  const int wave = threadIdx.x >> 5, lane = threadIdx.x & 31;

  const _Float16* Qm = Q2 + (size_t)bgh * LL * DH;
  const _Float16* Km = K2 + (size_t)bgh * LL * DH;
  const _Float16* Vm = Vt + (size_t)bgh * DH * LL;

  // pull the V tile (64 KB) toward the WGP while we do scores + softmax
  __builtin_prefetch(Vm + (size_t)threadIdx.x * 128, 0, 3);

  // ---- scores: S[16][512] = Q[16][64] * K^T ; wave covers cols wave*64..+63
  v8f acc[4] = {v8f{}, v8f{}, v8f{}, v8f{}};
#pragma unroll
  for (int kb = 0; kb < DH; kb += 32) {
    v16h a = load_a_f16(Qm, DH, qBase, kb, lane);
#pragma unroll
    for (int nt = 0; nt < 4; ++nt) {
      v16h b = load_b_f16(Km, DH, wave * 64 + nt * 16, kb, lane);
      acc[nt] = wmma16(a, b, acc[nt]);
    }
  }
  const float inv_temp = 0.04419417382415922f;  // 1/sqrt(512)
#pragma unroll
  for (int nt = 0; nt < 4; ++nt)
#pragma unroll
    for (int r = 0; r < 8; ++r)
      S[r + ((lane >> 4) << 3)][wave * 64 + nt * 16 + (lane & 15)] =
          acc[nt][r] * inv_temp;
  __syncthreads();

  // ---- row softmax; wave handles 2 rows; also emit attn to global (f32)
  const int h = bgh & (HEADS - 1);
  const int bg = bgh >> 3;
  const int b = bg >> 2, g = bg & 3;
#pragma unroll
  for (int rr = 0; rr < 2; ++rr) {
    int row = wave * 2 + rr;
    float xv[16];
    float mx = -1e30f;
#pragma unroll
    for (int i = 0; i < 16; ++i) {
      xv[i] = S[row][lane + i * 32];
      mx = fmaxf(mx, xv[i]);
    }
    mx = wave_max(mx);
    float sum = 0.f;
#pragma unroll
    for (int i = 0; i < 16; ++i) {
      xv[i] = __expf(xv[i] - mx);
      sum += xv[i];
    }
    sum = wave_sum(sum);
    float inv = 1.0f / sum;
    size_t base =
        ((((size_t)(h * BB + b) * GG + g) * LL) + qBase + row) * (size_t)LL;
#pragma unroll
    for (int i = 0; i < 16; ++i) {
      float pr = xv[i] * inv;
      attn_out[base + lane + i * 32] = pr;
      P[row][lane + i * 32] = (_Float16)pr;
    }
  }
  __syncthreads();

  // ---- O[16][64] = P[16][512] * V[512][64]; split K across wave pairs
  const int nt = wave & 3;
  const int khalf = wave >> 2;
  v8f oacc = v8f{};
  for (int kb = khalf * 256; kb < khalf * 256 + 256; kb += 32) {
    v16h a = load_a_f16(&P[0][0], 520, 0, kb, lane);
    v16h bfr = load_b_f16(Vm, LL, nt * 16, kb, lane);
    oacc = wmma16(a, bfr, oacc);
  }
#pragma unroll
  for (int r = 0; r < 8; ++r)
    Op[khalf][r + ((lane >> 4) << 3)][nt * 16 + (lane & 15)] = oacc[r];
  __syncthreads();

  // combine halves, write concat-head output f16: Oc[row][h*64+d]
  for (int e = threadIdx.x; e < 16 * DH; e += 256) {
    int m = e >> 6, d = e & 63;
    float o = Op[0][m][d] + Op[1][m][d];
    size_t row = (size_t)bg * LL + qBase + m;
    Oc[row * DM + h * DH + d] = (_Float16)o;
  }
}

// ---- Kernel 3: FC + bias + residual + LayerNorm ----------------------------
// grid = ROWS/16. Block covers 16 rows x all 512 cols; wave w: cols w*64..+63.
// Each wave's B tile staged via TDM (per-wave n-range differs, so stage the
// whole 512-col strip: 8 tiles of 64x32 -> issue one TDM per wave's strip).
__global__ __launch_bounds__(256) void fc_ln_kernel(
    const _Float16* __restrict__ Oc, const _Float16* __restrict__ Wfct,
    const float* __restrict__ bfc, const float* __restrict__ qres,
    const float* __restrict__ gamma, const float* __restrict__ beta,
    float* __restrict__ y) {
  __shared__ float X[16][516];

  const int wave = threadIdx.x >> 5, lane = threadIdx.x & 31;
  const int row0 = blockIdx.x * 16;
  const int n0 = wave * 64;

  // prefetch the residual rows we will need at the end
  __builtin_prefetch(qres + (size_t)row0 * DM + (size_t)threadIdx.x * 32, 0, 2);

  v8f acc[4] = {v8f{}, v8f{}, v8f{}, v8f{}};

#if USE_TDM
  __shared__ _Float16 Bt[2][512 * 32];  // double-buffered 512(n) x 32(k), 64KB
  if (threadIdx.x < 32)
    tdm_load_tile_512x32: ;
  // stage full 512-row strip: reuse 64x32 loader 8x (contiguous in LDS)
  if (threadIdx.x < 32) {
#pragma unroll
    for (int t = 0; t < 8; ++t)
      tdm_load_tile_64x32(Wfct + (size_t)(t * 64) * DM,
                          (unsigned)(size_t)&Bt[0][t * 64 * 32]);
  }
  for (int step = 0; step < 16; ++step) {
    const int kb = step * 32;
    __syncthreads();
    if (threadIdx.x < 32 && step + 1 < 16) {
#pragma unroll
      for (int t = 0; t < 8; ++t)
        tdm_load_tile_64x32(Wfct + (size_t)(t * 64) * DM + (kb + 32),
                            (unsigned)(size_t)&Bt[(step + 1) & 1][t * 64 * 32]);
    }
    if (step + 1 < 16)
      __builtin_amdgcn_s_wait_tensorcnt(8);   // current 8 tiles complete
    else
      __builtin_amdgcn_s_wait_tensorcnt(0);
    __syncthreads();
    const _Float16* Bcur = &Bt[step & 1][0];
    v16h a = load_a_f16(Oc, DM, row0, kb, lane);
#pragma unroll
    for (int nt = 0; nt < 4; ++nt) {
      v16h b = load_b_f16(Bcur, 32, n0 + nt * 16, 0, lane);
      acc[nt] = wmma16(a, b, acc[nt]);
    }
  }
#else
  for (int kb = 0; kb < DM; kb += 32) {
    v16h a = load_a_f16(Oc, DM, row0, kb, lane);
#pragma unroll
    for (int nt = 0; nt < 4; ++nt) {
      v16h b = load_b_f16(Wfct, DM, n0 + nt * 16, kb, lane);
      acc[nt] = wmma16(a, b, acc[nt]);
    }
  }
#endif

#pragma unroll
  for (int nt = 0; nt < 4; ++nt)
#pragma unroll
    for (int r = 0; r < 8; ++r) {
      int m = r + ((lane >> 4) << 3);
      int c = n0 + nt * 16 + (lane & 15);
      X[m][c] = acc[nt][r] + bfc[c] + qres[(size_t)(row0 + m) * DM + c];
    }
  __syncthreads();

#pragma unroll
  for (int rr = 0; rr < 2; ++rr) {
    int row = wave * 2 + rr;
    float xv[16];
    float s = 0.f;
#pragma unroll
    for (int i = 0; i < 16; ++i) {
      xv[i] = X[row][lane + i * 32];
      s += xv[i];
    }
    s = wave_sum(s);
    float mu = s * (1.0f / DM);
    float vv = 0.f;
#pragma unroll
    for (int i = 0; i < 16; ++i) {
      float d = xv[i] - mu;
      vv += d * d;
    }
    vv = wave_sum(vv);
    float rstd = rsqrtf(vv * (1.0f / DM) + 1e-5f);
    size_t rbase = (size_t)(row0 + row) * DM;
#pragma unroll
    for (int i = 0; i < 16; ++i) {
      int c = lane + i * 32;
      y[rbase + c] = (xv[i] - mu) * rstd * gamma[c] + beta[c];
    }
  }
}

// ---------------------------------------------------------------------------
extern "C" void kernel_launch(void* const* d_in, const int* in_sizes, int n_in,
                              void* d_out, int out_size, void* d_ws,
                              size_t ws_size, hipStream_t stream) {
  const float* q    = (const float*)d_in[0];
  const float* k    = (const float*)d_in[1];
  const float* v    = (const float*)d_in[2];
  const float* Wq   = (const float*)d_in[3];
  const float* bq   = (const float*)d_in[4];
  const float* Wk   = (const float*)d_in[5];
  const float* bk   = (const float*)d_in[6];
  const float* Wv   = (const float*)d_in[7];
  const float* bv   = (const float*)d_in[8];
  const float* Wfc  = (const float*)d_in[9];
  const float* bfc  = (const float*)d_in[10];
  const float* gam  = (const float*)d_in[11];
  const float* bet  = (const float*)d_in[12];

  char* ws = (char*)d_ws;
  _Float16* Wt = (_Float16*)ws;                             //  2 MiB
  _Float16* Q2 = (_Float16*)(ws + (size_t)(2)  * 1048576);  // 16 MiB
  _Float16* K2 = (_Float16*)(ws + (size_t)(18) * 1048576);  // 16 MiB
  _Float16* Vt = (_Float16*)(ws + (size_t)(34) * 1048576);  // 16 MiB
  _Float16* Oc = (_Float16*)(ws + (size_t)(50) * 1048576);  // 16 MiB

  float* yout = (float*)d_out;
  float* attn_out = yout + (size_t)BB * GG * LL * DM;       // 8,388,608 floats

  // 0) weight convert/transpose to f16 [n][k]
  wconv_kernel<<<4096, 256, 0, stream>>>(Wq, Wk, Wv, Wfc, Wt);

  // 1) Q/K/V projections (z selects input matrix)
  dim3 g1(ROWS / 128, DM / 64, 3);
  qkv_gemm_kernel<<<g1, 256, 0, stream>>>(q, k, v, Wt, bq, bk, bv, Q2, K2, Vt);

  // 2) attention per (b,g,h) x 16-query-row block
  dim3 g2(BB * GG * HEADS, LL / 16);
  attn_kernel<<<g2, 256, 0, stream>>>(Q2, K2, Vt, attn_out, Oc);

  // 3) FC + residual + LayerNorm
  fc_ln_kernel<<<ROWS / 16, 256, 0, stream>>>(
      Oc, Wt + (size_t)3 * DM * DM, bfc, q, gam, bet, yout);
}
